// GCconv2_8993661518261
// MI455X (gfx1250) — matrix operand order repaired and checked by reading, not compiled, not run on hardware
//
#include <hip/hip_runtime.h>
#include <hip/hip_bf16.h>
#include <math.h>

// ---------------------------------------------------------------------------
// Dynamic-filter ("GCconv2") pipeline for MI455X (gfx1250, wave32, WMMA).
//
//  weights = Wv @ (x @ softmax(Wk @ x)^T)   (V-GEMM eliminated by reassoc.)
//  out     = depthwise7x7(x, weights)
//
// GEMMs use v_wmma_f32_16x16x32_bf16. x is pre-converted to bf16 in BOTH
// [C,N] and transposed [N,C] layouts so every WMMA fragment load is a
// contiguous 16-byte run (global_load_b128) instead of stride-N b32 gathers.
// ---------------------------------------------------------------------------

typedef __attribute__((ext_vector_type(16))) __bf16 v16bf;
typedef __attribute__((ext_vector_type(8)))  float  v8f;

#define B_   32
#define C_   256
#define H_   56
#define W_   56
#define N_   (H_ * W_)     // 3136 spatial positions
#define KS_  7
#define KK_  49            // 7*7 filter taps
#define PAD_ 3

// ---------------------------------------------------------------------------
// Kernel 0: x (f32, [B,C,N]) -> xbf (bf16, [B,C,N]) and xT (bf16, [B,N,C]).
// 32x32 LDS-tiled transpose; all global accesses coalesced.
// ---------------------------------------------------------------------------
__global__ __launch_bounds__(256)
void k_cvt(const float* __restrict__ x, __bf16* __restrict__ xbf,
           __bf16* __restrict__ xT) {
  __shared__ __bf16 tile[32][33];              // +1 pad against bank conflicts
  const int b   = blockIdx.z;
  const int c0  = blockIdx.y * 32;
  const int n0  = blockIdx.x * 32;
  const int tid = threadIdx.x;

#pragma unroll
  for (int p = 0; p < 4; ++p) {
    const int idx = tid + p * 256;
    const int r = idx >> 5, col = idx & 31;    // r = local c, col = local n
    const float v = x[((size_t)b * C_ + c0 + r) * N_ + n0 + col];
    const __bf16 bv = (__bf16)v;
    tile[r][col] = bv;
    xbf[((size_t)b * C_ + c0 + r) * N_ + n0 + col] = bv;
  }
  __syncthreads();
#pragma unroll
  for (int p = 0; p < 4; ++p) {
    const int idx = tid + p * 256;
    const int r = idx >> 5, col = idx & 31;    // r = local n, col = local c
    xT[((size_t)b * N_ + n0 + r) * C_ + c0 + col] = tile[col][r];
  }
}

// ---------------------------------------------------------------------------
// Kernel 1: logits^T[n, kk] = sum_c xT[b, n, c] * Wk[kk, c]
// One wave per 16(n) x 64(kk padded) tile; K-dim C=256 in 8 steps of 32.
// A fragments: contiguous bf16 b128 runs from xT. Output written back into
// [kk, n] layout (8 consecutive floats per lane) for the softmax pass.
// ---------------------------------------------------------------------------
__global__ __launch_bounds__(32)
void k_logits(const __bf16* __restrict__ xT, const float* __restrict__ Wk,
              float* __restrict__ Lbuf) {
  const int lane = threadIdx.x & 31;
  const int half = lane >> 4;
  const int lid  = lane & 15;
  const int b  = blockIdx.y;
  const int n0 = blockIdx.x * 16;

  v8f acc[4];
#pragma unroll
  for (int t = 0; t < 4; ++t) { v8f z = {}; acc[t] = z; }

  const __bf16* xTb = xT + ((size_t)b * N_ + n0 + lid) * C_;

  for (int c0 = 0; c0 < C_; c0 += 32) {
    // A fragment: x^T tile [16 n x 32 c]; two contiguous 8-elem bf16 runs
    v16bf af;
    const __bf16* ap = xTb + c0 + half * 16;
#pragma unroll
    for (int e = 0; e < 8; ++e) af[e] = ap[e];
#pragma unroll
    for (int e = 0; e < 8; ++e) af[8 + e] = ap[8 + e];

#pragma unroll
    for (int kt = 0; kt < 4; ++kt) {
      // B fragment: Wk^T tile [32 c x 16 kk]; per-lane contiguous f32 run
      v16bf bfrag;
      const int kk = kt * 16 + lid;
      if (kk < KK_) {
        const float* bp = Wk + kk * C_ + c0 + half * 16;
#pragma unroll
        for (int e = 0; e < 16; ++e) bfrag[e] = (__bf16)bp[e];
      } else {
#pragma unroll
        for (int e = 0; e < 16; ++e) bfrag[e] = (__bf16)0.0f;
      }
      acc[kt] = __builtin_amdgcn_wmma_f32_16x16x32_bf16(
          false, af, false, bfrag, (short)0, acc[kt], false, false);
    }
  }

  // D layout: col kk = kt*16 + lid, row n = n0 + half*8 + v (v contiguous)
#pragma unroll
  for (int kt = 0; kt < 4; ++kt) {
    const int kk = kt * 16 + lid;
    if (kk < KK_) {
      float* op = Lbuf + ((size_t)b * KK_ + kk) * N_ + n0 + half * 8;
#pragma unroll
      for (int v = 0; v < 8; ++v) op[v] = acc[kt][v];
    }
  }
}

// ---------------------------------------------------------------------------
// Kernel 2: row softmax over N=3136 (one block per (b, kk) row), emit bf16.
// ---------------------------------------------------------------------------
__global__ __launch_bounds__(256)
void k_softmax(const float* __restrict__ Lbuf, __bf16* __restrict__ Kbuf) {
  __shared__ float red[256];
  const int row = blockIdx.x;                 // b*49 + kk
  const int tid = threadIdx.x;
  const float* src = Lbuf + (size_t)row * N_;

  float m = -1e30f;
  for (int i = tid; i < N_; i += 256) m = fmaxf(m, src[i]);
  red[tid] = m;
  __syncthreads();
  for (int s = 128; s > 0; s >>= 1) {
    if (tid < s) red[tid] = fmaxf(red[tid], red[tid + s]);
    __syncthreads();
  }
  m = red[0];
  __syncthreads();

  float sum = 0.0f;
  for (int i = tid; i < N_; i += 256) sum += __expf(src[i] - m);
  red[tid] = sum;
  __syncthreads();
  for (int s = 128; s > 0; s >>= 1) {
    if (tid < s) red[tid] += red[tid + s];
    __syncthreads();
  }
  const float inv = 1.0f / red[0];

  __bf16* dst = Kbuf + (size_t)row * N_;
  for (int i = tid; i < N_; i += 256)
    dst[i] = (__bf16)(__expf(src[i] - m) * inv);
}

// ---------------------------------------------------------------------------
// Kernel 3: S[b, c, kk] = sum_n xbf[b, c, n] * K[b, kk, n]
// Block = 4 waves splitting the 98 K-dim steps; LDS tree-combine; wave 0
// writes. All fragment loads are contiguous bf16 runs.
// ---------------------------------------------------------------------------
__global__ __launch_bounds__(128)
void k_corr(const __bf16* __restrict__ xbf, const __bf16* __restrict__ Kbuf,
            float* __restrict__ S) {
  __shared__ float part[4][4][32][8];  // [wave][coltile][lane][v] = 16 KB
  const int tid  = threadIdx.x;
  const int wave = tid >> 5;
  const int lane = tid & 31;
  const int half = lane >> 4;
  const int lid  = lane & 15;
  const int b     = blockIdx.y;
  const int ctile = blockIdx.x;        // 0..15 -> 16 channels each

  const __bf16* xb = xbf + ((size_t)b * C_ + ctile * 16 + lid) * N_;
  const __bf16* Kb = Kbuf + (size_t)b * KK_ * N_;

  v8f acc[4];
#pragma unroll
  for (int t = 0; t < 4; ++t) { v8f z = {}; acc[t] = z; }

  const int NSTEP = N_ / 32;           // 98
  for (int s = wave; s < NSTEP; s += 4) {
    const int n0 = s * 32;
    // A fragment: x tile [16 c x 32 n]; two contiguous 8-elem runs
    v16bf af;
    const __bf16* ap = xb + n0 + half * 16;
    __builtin_prefetch(ap + 128, 0, 1);          // next split-K chunk
#pragma unroll
    for (int e = 0; e < 8; ++e) af[e] = ap[e];
#pragma unroll
    for (int e = 0; e < 8; ++e) af[8 + e] = ap[8 + e];

#pragma unroll
    for (int ct = 0; ct < 4; ++ct) {
      // B fragment: K^T tile [32 n x 16 kk]; per-lane contiguous bf16 run
      v16bf bfrag;
      const int kkc = ct * 16 + lid;
      if (kkc < KK_) {
        const __bf16* bp = Kb + (size_t)kkc * N_ + n0 + half * 16;
#pragma unroll
        for (int e = 0; e < 16; ++e) bfrag[e] = bp[e];
      } else {
#pragma unroll
        for (int e = 0; e < 16; ++e) bfrag[e] = (__bf16)0.0f;
      }
      acc[ct] = __builtin_amdgcn_wmma_f32_16x16x32_bf16(
          false, af, false, bfrag, (short)0, acc[ct], false, false);
    }
  }

#pragma unroll
  for (int ct = 0; ct < 4; ++ct)
#pragma unroll
    for (int v = 0; v < 8; ++v) part[wave][ct][lane][v] = acc[ct][v];
  __syncthreads();

  if (wave == 0) {
#pragma unroll
    for (int ct = 0; ct < 4; ++ct) {
      const int kkc = ct * 16 + lid;
#pragma unroll
      for (int v = 0; v < 8; ++v) {
        const float sum = part[0][ct][lane][v] + part[1][ct][lane][v] +
                          part[2][ct][lane][v] + part[3][ct][lane][v];
        const int c = ctile * 16 + half * 8 + v;
        if (kkc < KK_)
          S[((size_t)b * C_ + c) * KK_ + kkc] = sum;
      }
    }
  }
}

// ---------------------------------------------------------------------------
// Kernel 4: weights[b, c, kk] = sum_c' Wv[c, c'] * S[b, c', kk]
// One wave per 16(c) x 64(kk padded) tile, K-dim C=256 in 8 steps.
// ---------------------------------------------------------------------------
__global__ __launch_bounds__(32)
void k_wgt(const float* __restrict__ Wv, const float* __restrict__ S,
           float* __restrict__ Wt) {
  const int lane = threadIdx.x & 31;
  const int half = lane >> 4;
  const int lid  = lane & 15;
  const int b     = blockIdx.y;
  const int ctile = blockIdx.x;        // 0..15

  const float* Sb = S + (size_t)b * C_ * KK_;

  v8f acc[4];
#pragma unroll
  for (int t = 0; t < 4; ++t) { v8f z = {}; acc[t] = z; }

  for (int c0 = 0; c0 < C_; c0 += 32) {
    // A fragment: Wv tile [16 c x 32 c']; per-lane contiguous f32 runs
    v16bf af;
    const float* ap = Wv + (ctile * 16 + lid) * C_ + c0 + half * 16;
#pragma unroll
    for (int e = 0; e < 8; ++e) af[e] = (__bf16)ap[e];
#pragma unroll
    for (int e = 0; e < 8; ++e) af[8 + e] = (__bf16)ap[8 + e];

#pragma unroll
    for (int ct = 0; ct < 4; ++ct) {
      // B fragment: S tile [32 c' x 16 kk]; cols >= 49 zero
      v16bf bfrag;
      const int kkc = ct * 16 + lid;
      if (kkc < KK_) {
#pragma unroll
        for (int e = 0; e < 16; ++e) {
          const int cp = c0 + half * 16 + e;
          bfrag[e] = (__bf16)Sb[(size_t)cp * KK_ + kkc];
        }
      } else {
#pragma unroll
        for (int e = 0; e < 16; ++e) bfrag[e] = (__bf16)0.0f;
      }
      acc[ct] = __builtin_amdgcn_wmma_f32_16x16x32_bf16(
          false, af, false, bfrag, (short)0, acc[ct], false, false);
    }
  }

#pragma unroll
  for (int ct = 0; ct < 4; ++ct) {
    const int kkc = ct * 16 + lid;
#pragma unroll
    for (int v = 0; v < 8; ++v) {
      const int c = ctile * 16 + half * 8 + v;
      if (kkc < KK_)
        Wt[((size_t)b * C_ + c) * KK_ + kkc] = acc[ct][v];
    }
  }
}

// ---------------------------------------------------------------------------
// Kernel 5: depthwise 7x7 conv, one (b, c) plane per block. 62x62 halo tile
// + 49-tap filter staged in LDS (15.6 KB of the 320 KB/WGP pool).
// ---------------------------------------------------------------------------
__global__ __launch_bounds__(256)
void k_dwconv(const float* __restrict__ x, const float* __restrict__ Wt,
              float* __restrict__ out) {
  __shared__ float sm[62 * 62];
  __shared__ float fw[KK_];
  const int tid = threadIdx.x;
  const int bc  = blockIdx.x;                    // b*C + c
  const float* xp = x + (size_t)bc * N_;

  if (tid < KK_) fw[tid] = Wt[(size_t)bc * KK_ + tid];

  for (int idx = tid; idx < 62 * 62; idx += 256) {
    const int gy = idx / 62, gx = idx - gy * 62;
    const int hy = gy - PAD_, hx = gx - PAD_;
    float v = 0.0f;
    if (hy >= 0 && hy < H_ && hx >= 0 && hx < W_) v = xp[hy * W_ + hx];
    sm[idx] = v;
  }
  __syncthreads();

  float* op = out + (size_t)bc * N_;
  for (int o = tid; o < N_; o += 256) {
    const int h = o / W_, w = o - h * W_;
    float sum = 0.0f;
#pragma unroll
    for (int i = 0; i < KS_; ++i)
#pragma unroll
      for (int j = 0; j < KS_; ++j)
        sum = fmaf(sm[(h + i) * 62 + (w + j)], fw[i * KS_ + j], sum);
    op[o] = sum;
  }
}

// ---------------------------------------------------------------------------
// Launcher.  Workspace layout (sequential-stream overlap is deterministic):
//   [ xT bf16 51.4MB | xbf bf16 51.4MB | Lbuf f32 19.7MB | S 1.6MB | Wt 1.6MB ]
//   Kbuf (9.8MB) overlays xT, which is dead after k_logits completes.
// ---------------------------------------------------------------------------
extern "C" void kernel_launch(void* const* d_in, const int* in_sizes, int n_in,
                              void* d_out, int out_size, void* d_ws,
                              size_t ws_size, hipStream_t stream) {
  const float* x  = (const float*)d_in[0];   // [B, C, H, W] f32
  const float* Wk = (const float*)d_in[1];   // [49, C]      f32
  const float* Wv = (const float*)d_in[2];   // [C, C]       f32
  float* out = (float*)d_out;                // [B, C, H, W] f32

  char* ws = (char*)d_ws;
  const size_t szT = (size_t)B_ * N_ * C_ * sizeof(__bf16);    // xT
  const size_t szX = (size_t)B_ * C_ * N_ * sizeof(__bf16);    // xbf
  const size_t szL = (size_t)B_ * KK_ * N_ * sizeof(float);    // logits
  const size_t szS = (size_t)B_ * C_ * KK_ * sizeof(float);    // x @ K^T

  __bf16* xT   = (__bf16*)(ws);
  __bf16* Kbuf = (__bf16*)(ws);                 // overlays xT (dead by then)
  __bf16* xbf  = (__bf16*)(ws + szT);
  float*  Lbuf = (float*)(ws + szT + szX);
  float*  S    = (float*)(ws + szT + szX + szL);
  float*  Wt   = (float*)(ws + szT + szX + szL + szS);

  // 0) f32 -> bf16 in straight + transposed layouts (one coalesced pass)
  k_cvt<<<dim3(N_ / 32, C_ / 32, B_), 256, 0, stream>>>(x, xbf, xT);
  // 1) logits^T = x^T @ Wk^T   (WMMA bf16, b128 fragment loads)
  k_logits<<<dim3(N_ / 16, B_), 32, 0, stream>>>(xT, Wk, Lbuf);
  // 2) softmax over spatial dim, emit bf16 K  (Kbuf reuses xT space)
  k_softmax<<<B_ * KK_, 256, 0, stream>>>(Lbuf, Kbuf);
  // 3) S = x @ K^T             (WMMA bf16, 4-wave split-K over N)
  k_corr<<<dim3(16, B_), 128, 0, stream>>>(xbf, Kbuf, S);
  // 4) weights = Wv @ S        (WMMA bf16)
  k_wgt<<<dim3(16, B_), 32, 0, stream>>>(Wv, S, Wt);
  // 5) depthwise 7x7 conv with dynamic per-(b,c) filters
  k_dwconv<<<B_ * C_, 256, 0, stream>>>(x, Wt, out);
}
